// AutoregressiveSplineDeep_27041114096366
// MI455X (gfx1250) — compile-verified, matrix operand order checked by reading, not run on hardware
//
#include <hip/hip_runtime.h>

// ---------------------------------------------------------------------------
// AutoregressiveSplineDeep on gfx1250 (MI455X)
//   MADE mask structure (INPUT_DIM=2) collapses to:
//     h1 = relu(z0 * W1[:,0] + b1)                (rank-1, VALU)
//     h2 = relu(h1 @ W2^T + b2)                   (256x256 -> WMMA f16)
//     raw = h2 @ W3[odd rows]^T + b3[odd]         (64x256  -> WMMA f16)
//     dim-0 spline params == b3[even] (constant, precomputed once)
//   Weights staged into padded-row LDS via TDM (tensor_load_to_lds) when the
//   builtin exists; the D# pad fields (interval=128 dwords, amount=4 dwords)
//   reproduce the 528-byte row pitch exactly.
// ---------------------------------------------------------------------------

#define NSAMP   131072
#define HID     256
#define KB      16
#define MIN_BIN 1e-3f
#define MIN_DER 1e-3f
#define MIN_LAM 0.025f
#define BOUND_F 5.0f

typedef __attribute__((ext_vector_type(16))) _Float16 v16h;
typedef __attribute__((ext_vector_type(8)))  _Float16 h8;
typedef __attribute__((ext_vector_type(4)))  _Float16 h4;
typedef __attribute__((ext_vector_type(8)))  float    v8f;
typedef __attribute__((ext_vector_type(4)))  unsigned int v4u;
typedef __attribute__((ext_vector_type(8)))  int      v8i;
typedef __attribute__((ext_vector_type(4)))  int      v4i;

// LDS layout (bytes); padded f16 row stride = 264 halves (528 B, 16B aligned,
// staggers rows by 4 banks -> conflict-free b128 tile loads)
#define PADH      264
#define OWS       66           // out_raw f32 row stride
#define W2S_OFF   0            // 256*264*2 = 135168
#define W3S_OFF   135168       //  64*264*2 =  33792
#define H1S_OFF   168960       // 128*264*2 =  67584
#define H2S_OFF   236544       // 128*264*2 =  67584
#define SMEM_BYTES 304128      // < 320KB per-WGP LDS

#if defined(__has_builtin)
#if __has_builtin(__builtin_amdgcn_tensor_load_to_lds)
#define HAVE_TDM 1
#endif
#endif
#ifndef HAVE_TDM
#define HAVE_TDM 0
#endif

// ---------------- spline helpers (monotonic linear rational spline) --------

__device__ __forceinline__ float softplus_(float x) {
  return (x > 20.f) ? x : log1pf(expf(x));
}

__device__ void build_lrs(const float raw[63], float cw[17], float ch[17],
                          float der[17], float lam[16]) {
  const float AA = 1.0f - MIN_BIN * (float)KB;
  {
    float m = raw[0];
#pragma unroll
    for (int k = 1; k < 16; ++k) m = fmaxf(m, raw[k]);
    float e[16]; float s = 0.f;
#pragma unroll
    for (int k = 0; k < 16; ++k) { e[k] = expf(raw[k] - m); s += e[k]; }
    float inv = 1.0f / s, cum = 0.f;
    cw[0] = -BOUND_F;
#pragma unroll
    for (int k = 0; k < 16; ++k) {
      cum += MIN_BIN + AA * e[k] * inv;
      cw[k + 1] = -BOUND_F + 2.f * BOUND_F * cum;
    }
    cw[16] = BOUND_F;
  }
  {
    float m = raw[16];
#pragma unroll
    for (int k = 1; k < 16; ++k) m = fmaxf(m, raw[16 + k]);
    float e[16]; float s = 0.f;
#pragma unroll
    for (int k = 0; k < 16; ++k) { e[k] = expf(raw[16 + k] - m); s += e[k]; }
    float inv = 1.0f / s, cum = 0.f;
    ch[0] = -BOUND_F;
#pragma unroll
    for (int k = 0; k < 16; ++k) {
      cum += MIN_BIN + AA * e[k] * inv;
      ch[k + 1] = -BOUND_F + 2.f * BOUND_F * cum;
    }
    ch[16] = BOUND_F;
  }
  der[0] = 1.f; der[16] = 1.f;
#pragma unroll
  for (int j = 0; j < 15; ++j) der[j + 1] = MIN_DER + softplus_(raw[32 + j]);
#pragma unroll
  for (int k = 0; k < 16; ++k)
    lam[k] = MIN_LAM + (1.f - 2.f * MIN_LAM) / (1.f + expf(-raw[47 + k]));
}

__device__ float lrs_eval(float x, const float cw[17], const float ch[17],
                          const float der[17], const float lam[16]) {
  bool inside = (x >= -BOUND_F) && (x <= BOUND_F);
  float xc = fminf(fmaxf(x, -BOUND_F), BOUND_F);
  int bin = 0;
#pragma unroll
  for (int k = 1; k <= 15; ++k) bin += (xc >= cw[k]) ? 1 : 0;
  float xk = cw[0], xk1 = cw[1], yk = ch[0], yk1 = ch[1];
  float dk = der[0], dk1 = der[1], lk = lam[0];
#pragma unroll
  for (int k = 1; k < 16; ++k) {
    if (bin == k) {
      xk = cw[k]; xk1 = cw[k + 1]; yk = ch[k]; yk1 = ch[k + 1];
      dk = der[k]; dk1 = der[k + 1]; lk = lam[k];
    }
  }
  float wk = xk1 - xk, hk = yk1 - yk;
  float wb = sqrtf(dk / dk1);
  float wc = (lk * dk + (1.f - lk) * wb * dk1) * (wk / hk);
  float ya = yk, yb = yk1;
  float yc = ((1.f - lk) * ya + lk * wb * yb) / ((1.f - lk) + lk * wb);
  float th = (xc - xk) / wk;
  bool ls = th <= lk;
  float num = ls ? (ya * (lk - th) + wc * yc * th)
                 : (wc * yc * (1.f - th) + wb * yb * (th - lk));
  float den = ls ? ((lk - th) + wc * th)
                 : (wc * (1.f - th) + wb * (th - lk));
  float y = num / den;
  return inside ? y : x;
}

// ---------------- WMMA tile loader (ISA 16-bit A/B operand layout) ---------
// lane L holds row/col L%16; element i -> K = 16*(i>=8) + 8*(L>=16) + (i%8)
__device__ __forceinline__ v16h ldtile(const _Float16* p) {
  h8 lo = *(const h8*)p;
  h8 hi = *(const h8*)(p + 16);
  v16h r;
#pragma unroll
  for (int i = 0; i < 8; ++i) { r[i] = lo[i]; r[i + 8] = hi[i]; }
  return r;
}

// ---------------- TDM: 2-D f16 tile -> padded-row LDS ----------------------
// rows x 256 f16, row pitch in LDS = 528 B via D# pad (128 dwords + 4 dwords)
#if HAVE_TDM
__device__ __forceinline__ void tdm_load_2d_f16(const _Float16* gsrc,
                                                unsigned lds_off,
                                                unsigned rows) {
  unsigned long long ga = (unsigned long long)(const void*)gsrc;
  v4u g0;
  g0[0] = 1u;                                            // count=1, user desc
  g0[1] = lds_off;                                       // lds_addr (bytes)
  g0[2] = (unsigned)(ga & 0xffffffffu);                  // global_addr[31:0]
  g0[3] = (unsigned)((ga >> 32) & 0x01ffffffu) | 0x80000000u; // [56:32]|type=2
  v8i g1;
  g1[0] = (int)((1u << 16)    // data_size = 2 bytes
              | (1u << 20)    // pad_enable
              | (6u << 22)    // pad_interval: 128 dwords (= 512B row)
              | (3u << 25));  // pad_amount:   4 dwords (= 16B pad)
  g1[1] = (int)(256u << 16);  // tensor_dim0[15:0] = 256 (bits 79:48)
  g1[2] = (int)(rows << 16);  // tensor_dim0 hi=0 | tensor_dim1[15:0]=rows
  g1[3] = (int)(256u << 16);  // tensor_dim1 hi=0 | tile_dim0 = 256
  g1[4] = (int)rows;          // tile_dim1 = rows | tile_dim2 = 0
  g1[5] = 256;                // tensor_dim0_stride[31:0] = 256 elements
  g1[6] = 0;                  // stride hi | tensor_dim1_stride lo
  g1[7] = 0;
  v4i z = {0, 0, 0, 0};
#if __clang_major__ >= 23
  v8i z8 = {0, 0, 0, 0, 0, 0, 0, 0};
  __builtin_amdgcn_tensor_load_to_lds(g0, g1, z, z, z8, 0);
#else
  __builtin_amdgcn_tensor_load_to_lds(g0, g1, z, z, 0);
#endif
}
#endif

// ---------------- setup kernels --------------------------------------------

__global__ void k_prep_weights(const float* __restrict__ W1,
                               const float* __restrict__ W2,
                               const float* __restrict__ W3,
                               _Float16* __restrict__ w2h,
                               _Float16* __restrict__ w3h,
                               float* __restrict__ w1c) {
  int i = blockIdx.x * 256 + threadIdx.x;   // grid = 320 blocks -> 81920
  int row = i >> 8, col = i & 255;
  if (row < 256) {
    w2h[i] = (_Float16)W2[i];
    if (col == 0) w1c[row] = W1[row * 2];   // masked column 0 of W1
  } else {
    int r = row - 256;                      // packed odd rows of W3, row 63 = 0
    w3h[r * 256 + col] = (r < 63) ? (_Float16)W3[(2 * r + 1) * 256 + col]
                                  : (_Float16)0.f;
  }
}

__global__ void k_prep_spline0(const float* __restrict__ b3,
                               float* __restrict__ sp0) {
  if (threadIdx.x != 0 || blockIdx.x != 0) return;
  float raw[63];
#pragma unroll
  for (int i = 0; i < 63; ++i) raw[i] = b3[2 * i];   // even rows: bias only
  float cw[17], ch[17], der[17], lam[16];
  build_lrs(raw, cw, ch, der, lam);
#pragma unroll
  for (int i = 0; i < 17; ++i) { sp0[i] = cw[i]; sp0[17 + i] = ch[i]; sp0[34 + i] = der[i]; }
#pragma unroll
  for (int i = 0; i < 16; ++i) sp0[51 + i] = lam[i];
}

__global__ void k_copy(const float* __restrict__ src, float* __restrict__ dst) {
  int i = blockIdx.x * 256 + threadIdx.x;
  dst[i] = src[i];
}

// ---------------- one flow step --------------------------------------------
// 256 threads = 8 waves; 128 samples/WG; wave w owns M-tile (samples 16w..16w+15)

__global__ __launch_bounds__(256, 1) void k_flow_step(
    const float* __restrict__ zin, float* __restrict__ zout,
    const _Float16* __restrict__ w2h, const _Float16* __restrict__ w3h,
    const float* __restrict__ w1c, const float* __restrict__ b1,
    const float* __restrict__ b2, const float* __restrict__ b3,
    const float* __restrict__ sp0) {
  extern __shared__ char smem[];
  _Float16* W2s = (_Float16*)(smem + W2S_OFF);
  _Float16* W3s = (_Float16*)(smem + W3S_OFF);
  _Float16* H1s = (_Float16*)(smem + H1S_OFF);
  _Float16* H2s = (_Float16*)(smem + H2S_OFF);
  float*    OutR = (float*)smem;               // aliases W2s after barrier

  const int tid = threadIdx.x;
  const int base = blockIdx.x * 128;

#if HAVE_TDM
  // DMA the padded-row weight tiles straight into LDS (TENSORcnt tracked),
  // overlapping with the rank-1 layer-1 compute below.
  if (tid < 32)       tdm_load_2d_f16(w2h, W2S_OFF, 256);
  else if (tid < 64)  tdm_load_2d_f16(w3h, W3S_OFF, 64);
#endif

  // layer 1: h1 = relu(z0 * w1c + b1); thread t -> sample t&127, k-half t>>7
  {
    int s = tid & 127, half = tid >> 7;
    float z0 = zin[(base + s) * 2];
    int k0 = half * 128;
#pragma unroll 4
    for (int k = k0; k < k0 + 128; k += 4) {
      float4 wv = *(const float4*)(w1c + k);
      float4 bv = *(const float4*)(b1 + k);
      h4 hv;
      hv[0] = (_Float16)fmaxf(0.f, fmaf(z0, wv.x, bv.x));
      hv[1] = (_Float16)fmaxf(0.f, fmaf(z0, wv.y, bv.y));
      hv[2] = (_Float16)fmaxf(0.f, fmaf(z0, wv.z, bv.z));
      hv[3] = (_Float16)fmaxf(0.f, fmaf(z0, wv.w, bv.w));
      *(h4*)(H1s + s * PADH + k) = hv;
    }
  }

#if HAVE_TDM
  if (tid < 64) __builtin_amdgcn_s_wait_tensorcnt(0);
#else
  // fallback: stage f16 weights into LDS with padded rows (b128 copies)
  __builtin_prefetch(w2h, 0, 3);
#pragma unroll 1
  for (int u = tid; u < 320 * 32; u += 256) {
    int r = u >> 5, s = u & 31;
    const h8* src = (r < 256) ? (const h8*)(w2h + r * 256 + s * 8)
                              : (const h8*)(w3h + (r - 256) * 256 + s * 8);
    h8* dst = (r < 256) ? (h8*)(W2s + r * PADH + s * 8)
                        : (h8*)(W3s + (r - 256) * PADH + s * 8);
    *dst = *src;
  }
#endif
  __syncthreads();

  const int wave = tid >> 5, lane = tid & 31;
  const int lm = lane & 15, lhi = lane >> 4;

  // layer 2: H2 = relu(H1 @ W2^T + b2)   [128 WMMAs/wave, A reg-cached]
  v16h A[8];
  {
    const _Float16* arow = H1s + (wave * 16 + lm) * PADH;
#pragma unroll
    for (int kt = 0; kt < 8; ++kt) A[kt] = ldtile(arow + kt * 32 + lhi * 8);
  }
#pragma unroll 1
  for (int nt = 0; nt < 16; ++nt) {
    v8f c = {};
    const _Float16* brow = W2s + (nt * 16 + lm) * PADH;
#pragma unroll
    for (int kt = 0; kt < 8; ++kt) {
      v16h b = ldtile(brow + kt * 32 + lhi * 8);
      c = __builtin_amdgcn_wmma_f32_16x16x32_f16(false, A[kt], false, b,
                                                 (short)0, c, false, false);
    }
    float bias = b2[nt * 16 + lm];
#pragma unroll
    for (int j = 0; j < 8; ++j) {
      float v = fmaxf(0.f, c[j] + bias);
      H2s[(wave * 16 + j + lhi * 8) * PADH + nt * 16 + lm] = (_Float16)v;
    }
  }
  __syncthreads();   // all waves done with W2s -> safe to alias as OutR

  // layer 3: raw = H2 @ W3odd^T          [32 WMMAs/wave]
  {
    const _Float16* arow = H2s + (wave * 16 + lm) * PADH;
#pragma unroll
    for (int kt = 0; kt < 8; ++kt) A[kt] = ldtile(arow + kt * 32 + lhi * 8);
  }
#pragma unroll 1
  for (int nt = 0; nt < 4; ++nt) {
    v8f c = {};
    const _Float16* brow = W3s + (nt * 16 + lm) * PADH;
#pragma unroll
    for (int kt = 0; kt < 8; ++kt) {
      v16h b = ldtile(brow + kt * 32 + lhi * 8);
      c = __builtin_amdgcn_wmma_f32_16x16x32_f16(false, A[kt], false, b,
                                                 (short)0, c, false, false);
    }
#pragma unroll
    for (int j = 0; j < 8; ++j)
      OutR[(wave * 16 + j + lhi * 8) * OWS + nt * 16 + lm] = c[j];
  }
  __syncthreads();

  // spline: thread t -> sample t>>1, dim t&1
  {
    int sId = tid >> 1, dim = tid & 1;
    int gs = base + sId;
    float x = zin[gs * 2 + dim];
    float cw[17], ch[17], der[17], lam[16];
    if (dim == 0) {                 // constant params precomputed from b3[even]
#pragma unroll
      for (int i = 0; i < 17; ++i) { cw[i] = sp0[i]; ch[i] = sp0[17 + i]; der[i] = sp0[34 + i]; }
#pragma unroll
      for (int i = 0; i < 16; ++i) lam[i] = sp0[51 + i];
    } else {                        // params from MADE head + b3[odd]
      float raw[63];
      const float* rowp = OutR + sId * OWS;
#pragma unroll
      for (int i = 0; i < 63; ++i) raw[i] = rowp[i] + b3[2 * i + 1];
      build_lrs(raw, cw, ch, der, lam);
    }
    zout[gs * 2 + dim] = lrs_eval(x, cw, ch, der, lam);
  }
}

// ---------------- host entry ------------------------------------------------

extern "C" void kernel_launch(void* const* d_in, const int* in_sizes, int n_in,
                              void* d_out, int out_size, void* d_ws, size_t ws_size,
                              hipStream_t stream) {
  (void)in_sizes; (void)n_in; (void)out_size; (void)ws_size;
  const float* x  = (const float*)d_in[0];
  const float* W1 = (const float*)d_in[1];
  const float* b1 = (const float*)d_in[2];
  const float* W2 = (const float*)d_in[3];
  const float* b2 = (const float*)d_in[4];
  const float* W3 = (const float*)d_in[5];
  const float* b3 = (const float*)d_in[6];
  float* out = (float*)d_out;
  char* ws = (char*)d_ws;
  _Float16* w2h = (_Float16*)(ws + 0);        // 256*256 f16
  _Float16* w3h = (_Float16*)(ws + 131072);   //  64*256 f16
  float*    w1c = (float*)(ws + 163840);      // 256 f32
  float*    sp0 = (float*)(ws + 164864);      // 67 f32

  hipFuncSetAttribute((const void*)k_flow_step,
                      hipFuncAttributeMaxDynamicSharedMemorySize, SMEM_BYTES);

  k_prep_weights<<<320, 256, 0, stream>>>(W1, W2, W3, w2h, w3h, w1c);
  k_prep_spline0<<<1, 64, 0, stream>>>(b3, sp0);
  k_copy<<<NSAMP * 2 / 256, 256, 0, stream>>>(x, out);   // zs[0] = x
  for (int step = 0; step < 4; ++step) {
    const float* zi = out + (size_t)step * NSAMP * 2;
    float*       zo = out + (size_t)(step + 1) * NSAMP * 2;
    k_flow_step<<<NSAMP / 128, 256, SMEM_BYTES, stream>>>(
        zi, zo, w2h, w3h, w1c, b1, b2, b3, sp0);
  }
}
